// BSLRecognitionModel_74577812128176
// MI455X (gfx1250) — matrix operand order repaired and checked by reading, not compiled, not run on hardware
//
#include <hip/hip_runtime.h>
#include <hip/hip_bf16.h>

typedef _Float16 f16;
typedef __attribute__((ext_vector_type(16))) _Float16 v16h;
typedef __attribute__((ext_vector_type(8)))  _Float16 f16x8;
typedef __attribute__((ext_vector_type(8)))  float    v8f;
typedef __attribute__((ext_vector_type(4)))  unsigned int u32x4;
typedef __attribute__((ext_vector_type(8)))  int      i32x8;
typedef __attribute__((ext_vector_type(4)))  int      i32x4;

#define BB 8192
#define TT 30
#define NBT (BB * TT)   // 245760

#if __has_builtin(__builtin_amdgcn_tensor_load_to_lds) && __has_builtin(__builtin_amdgcn_s_wait_tensorcnt)
  #define HAVE_TDM 1
  #if __has_include(<hip/amd_detail/amd_gfx1250_TDM.h>)
    #define TDM6 1   // amdgpu-toolchain: 6-arg builtin
  #else
    #define TDM6 0   // ROCm 7.2: 5-arg builtin
  #endif
#else
  #define HAVE_TDM 0
#endif

__device__ __forceinline__ float sigf(float x) {
    return __fdividef(1.0f, 1.0f + __expf(-x));
}

// ---------------------------------------------------------------------------
// CDNA5 wave32 WMMA fragment loaders (CDNA5 ISA §7.12.2), fully vectorized.
// A 16x32 f16 (MxK), lane m = lane%16, group = lane/16:
//   elem e -> K = kb + 16*group + e          (32B contiguous)
// B 32x16 f16 (KxN), lane n = lane%16, group = lane/16:
//   elem e -> K = kb + 16*group + e          (32B contiguous in a K-major row)
// C/D f32: vgpr v -> M = v + 8*group, N = lane%16
// ---------------------------------------------------------------------------
template <typename P>
__device__ __forceinline__ v16h frag_a16(const P* p /* row + kb */, int group) {
    f16x8 lo = *(const f16x8*)(p + 16 * group);
    f16x8 hi = *(const f16x8*)(p + 16 * group + 8);
    v16h a;
#pragma unroll
    for (int e = 0; e < 8; ++e) { a[e] = lo[e]; a[e + 8] = hi[e]; }
    return a;
}

template <typename P>
__device__ __forceinline__ v16h frag_b16(const P* p /* wrowT + kb + 16*group */) {
    f16x8 lo = *(const f16x8*)(p);
    f16x8 hi = *(const f16x8*)(p + 8);
    v16h b;
#pragma unroll
    for (int e = 0; e < 8; ++e) { b[e] = lo[e]; b[e + 8] = hi[e]; }
    return b;
}

#if HAVE_TDM
// TDM: contiguous 2D tile load Global->LDS. D# per CDNA5 ISA.
// group0: count=1 | lds_addr | global_addr[56:0] | type=2
// group1: data_size=2B | tensor_dim0=width | tensor_dim1=height |
//         tile_dim0=width | tile_dim1=height | tensor_dim0_stride=width
__device__ __forceinline__ void tdm_load_2d(unsigned lds_off, const void* gptr,
                                            unsigned width, unsigned height) {
    unsigned long long ga = (unsigned long long)gptr;
    u32x4 g0;
    g0.x = 1u;                                   // count=1 (valid descriptor)
    g0.y = lds_off;                              // LDS byte address
    g0.z = (unsigned)(ga & 0xffffffffu);         // global_addr[31:0]
    g0.w = (unsigned)((ga >> 32) & 0x01ffffffu)  // global_addr[56:32]
         | 0x80000000u;                          // type=2 ("image")
    i32x8 g1;
    g1[0] = (int)(1u << 16);                                        // data_size=1 (2 bytes)
    g1[1] = (int)((width & 0xffffu) << 16);                         // tensor_dim0[15:0]
    g1[2] = (int)(((width >> 16) & 0xffffu) | ((height & 0xffffu) << 16)); // dim0 hi | dim1 lo
    g1[3] = (int)(((height >> 16) & 0xffffu) | ((width & 0xffffu) << 16)); // dim1 hi | tile_dim0
    g1[4] = (int)(height & 0xffffu);                                // tile_dim1
    g1[5] = (int)width;                                             // tensor_dim0_stride lo32
    g1[6] = 0;
    g1[7] = 0;
    i32x4 z4 = {0, 0, 0, 0};
#if TDM6
    i32x8 z8 = {0, 0, 0, 0, 0, 0, 0, 0};
    __builtin_amdgcn_tensor_load_to_lds(g0, g1, z4, z4, z8, 0);
#else
    __builtin_amdgcn_tensor_load_to_lds(g0, g1, z4, z4, 0);
#endif
}
#endif

// ---------------------------------------------------------------------------
// WMMA GEMM: C[MxN] = act(A[MxK]_f16 @ B + bias), B given TRANSPOSED (N x K).
// One wave per block; wave computes a 16 x (16*NT) strip, A frag reused NT x.
// ---------------------------------------------------------------------------
template <int NT, bool RELU, bool OUT16>
__global__ void wmma_gemm_bias(const f16* __restrict__ A, const f16* __restrict__ BT,
                               const float* __restrict__ bias,
                               f16* __restrict__ C16, float* __restrict__ C32,
                               int M, int N, int K) {
    const int lane  = threadIdx.x & 31;
    const int group = lane >> 4;
    const int nl    = lane & 15;
    const int m0    = blockIdx.x << 4;
    const int n0    = blockIdx.y * (16 * NT);
    const f16* arow = A + (long)(m0 + nl) * K;
    v8f acc[NT];
#pragma unroll
    for (int j = 0; j < NT; ++j) acc[j] = (v8f){};
    for (int kb = 0; kb < K; kb += 32) {
        v16h af = frag_a16(arow + kb, group);
#pragma unroll
        for (int j = 0; j < NT; ++j) {
            const f16* brow = BT + (long)(n0 + j * 16 + nl) * K;
            v16h bf = frag_b16(brow + kb + 16 * group);
            acc[j] = __builtin_amdgcn_wmma_f32_16x16x32_f16(false, af, false, bf,
                                                            (short)0, acc[j], false, false);
        }
    }
#pragma unroll
    for (int j = 0; j < NT; ++j) {
#pragma unroll
        for (int v = 0; v < 8; ++v) {
            int row = m0 + v + 8 * group;
            int col = n0 + j * 16 + nl;
            float val = acc[j][v] + (bias ? bias[col] : 0.0f);
            if (RELU) val = fmaxf(val, 0.0f);
            if (OUT16) C16[(long)row * N + col] = (f16)val;
            else       C32[(long)row * N + col] = val;
        }
    }
}

// ---------------------------------------------------------------------------
// Fused BiLSTM direction scan. Block owns 16 batch rows, loops all T steps.
// z_t = [x_t , h_{t-1}] @ Wcat + b via WMMA. Weights transposed (N4 x KTOT);
// STAGE=true pulls them into LDS once via the Tensor Data Mover.
// ---------------------------------------------------------------------------
template <int DINP, int U, bool STAGE>
__global__ void lstm_scan(const f16* __restrict__ xin,     // (B,T,DINP) f16
                          const f16* __restrict__ wcatT,   // (4U, DINP+U) f16
                          const float* __restrict__ bias,  // (4U)
                          f16* __restrict__ out,           // (B,T,2U) f16
                          int reverse, int dircol) {
    constexpr int KTOT   = DINP + U;
    constexpr int N4     = 4 * U;
    constexpr int NTILES = N4 / 16;
    constexpr int NW     = 8;             // waves per block (blockDim = 256)
    constexpr int TPW    = NTILES / NW;   // N-tiles per wave
    static_assert(NTILES % NW == 0, "tile split");

    extern __shared__ char smem[];
    f16*   wlds = (f16*)smem;             // staged weights (STAGE only), offset 0
    char*  dyn  = smem + (STAGE ? (size_t)N4 * KTOT * sizeof(f16) : 0);
    f16*   hbuf = (f16*)dyn;                                     // 16*U f16
    float* cbuf = (float*)(dyn + (size_t)16 * U * sizeof(f16));  // 16*U f32
    float* zbuf = cbuf + 16 * U;                                 // 16*N4 f32

    const int tid   = threadIdx.x;
    const int lane  = tid & 31;
    const int wave  = tid >> 5;
    const int group = lane >> 4;
    const int ml    = lane & 15;
    const int b0    = blockIdx.x << 4;

    if constexpr (STAGE) {
#if HAVE_TDM
        if (tid == 0) {
            // dynamic-LDS base == 0 (no static LDS in this kernel)
            tdm_load_2d(0u, wcatT, KTOT, N4);
            __builtin_amdgcn_s_wait_tensorcnt(0);
        }
#else
        for (int i = tid; i < (N4 * KTOT) / 8; i += blockDim.x)
            ((f16x8*)wlds)[i] = ((const f16x8*)wcatT)[i];
#endif
    }
    for (int i = tid; i < 16 * U; i += blockDim.x) { hbuf[i] = (f16)0.0f; cbuf[i] = 0.0f; }
    __syncthreads();

    const f16* hrow = hbuf + ml * U;

    auto run = [&](const auto* wsrc) {
        v8f acc[TPW];
        for (int t = 0; t < TT; ++t) {
            const int tt = reverse ? (TT - 1 - t) : t;
            const f16* xrow = xin + ((long)(b0 + ml) * TT + tt) * DINP;
#pragma unroll
            for (int j = 0; j < TPW; ++j) acc[j] = (v8f){};

            // K-chunks from x_t (global)
            for (int kb = 0; kb < DINP; kb += 32) {
                v16h af = frag_a16(xrow + kb, group);
#pragma unroll
                for (int j = 0; j < TPW; ++j) {
                    const auto* brow = wsrc + (long)((wave + j * NW) * 16 + ml) * KTOT;
                    v16h bf = frag_b16(brow + kb + 16 * group);
                    acc[j] = __builtin_amdgcn_wmma_f32_16x16x32_f16(false, af, false, bf,
                                                                    (short)0, acc[j], false, false);
                }
            }
            // K-chunks from h_{t-1} (LDS)
            for (int kb = 0; kb < U; kb += 32) {
                v16h af = frag_a16(hrow + kb, group);
#pragma unroll
                for (int j = 0; j < TPW; ++j) {
                    const auto* brow = wsrc + (long)((wave + j * NW) * 16 + ml) * KTOT;
                    v16h bf = frag_b16(brow + DINP + kb + 16 * group);
                    acc[j] = __builtin_amdgcn_wmma_f32_16x16x32_f16(false, af, false, bf,
                                                                    (short)0, acc[j], false, false);
                }
            }
#pragma unroll
            for (int j = 0; j < TPW; ++j) {
                const int col = (wave + j * NW) * 16 + ml;
#pragma unroll
                for (int v = 0; v < 8; ++v)
                    zbuf[(v + 8 * group) * N4 + col] = acc[j][v] + bias[col];
            }
            __syncthreads();

            // gates: i,f,g,o contiguous U-chunks along the 4U axis
            for (int i = tid; i < 16 * U; i += blockDim.x) {
                int m = i / U, j = i - m * U;
                const float* zr = zbuf + m * N4;
                float zi = zr[j], zf = zr[U + j], zg = zr[2 * U + j], zo = zr[3 * U + j];
                float c = cbuf[i];
                c = sigf(zf) * c + sigf(zi) * fmaxf(zg, 0.0f);
                float h = sigf(zo) * fmaxf(c, 0.0f);
                cbuf[i] = c;
                hbuf[i] = (f16)h;
                out[((long)(b0 + m) * TT + tt) * (2 * U) + dircol * U + j] = (f16)h;
            }
            __syncthreads();
        }
    };
    if constexpr (STAGE) run(wlds); else run(wcatT);
}

// ---------------------------------------------------------------------------
// Data-prep / epilogue kernels
// ---------------------------------------------------------------------------
__global__ void pad_convert(const float* __restrict__ s, f16* __restrict__ d,
                            int rows, int cols, int colsp) {
    long i = (long)blockIdx.x * blockDim.x + threadIdx.x;
    long n = (long)rows * colsp;
    if (i >= n) return;
    int r = (int)(i / colsp), c = (int)(i - (long)r * colsp);
    d[i] = (c < cols) ? (f16)s[(long)r * cols + c] : (f16)0.0f;
}

// transpose-convert: src (K x N) f32  ->  dst[n*ld + colOff + k] f16
__global__ void convT_f16(const float* __restrict__ s, f16* __restrict__ d,
                          int K, int N, int ld, int colOff) {
    long i = (long)blockIdx.x * blockDim.x + threadIdx.x;
    long total = (long)K * N;
    if (i >= total) return;
    int k = (int)(i / N), n = (int)(i - (long)k * N);
    d[(long)n * ld + colOff + k] = (f16)s[i];
}

__global__ void zero_f16(f16* __restrict__ d, long n) {
    long i = (long)blockIdx.x * blockDim.x + threadIdx.x;
    if (i < n) d[i] = (f16)0.0f;
}

__global__ void bn_kernel(const f16* __restrict__ x, const float* __restrict__ g,
                          const float* __restrict__ b, const float* __restrict__ m,
                          const float* __restrict__ v, f16* __restrict__ y, long total) {
    long i = (long)blockIdx.x * blockDim.x + threadIdx.x;
    if (i >= total) return;
    int j = (int)(i & 63);
    float xv = (float)x[i];
    y[i] = (f16)((xv - m[j]) * rsqrtf(v[j] + 1e-3f) * g[j] + b[j]);
}

__global__ void softmax26(const float* __restrict__ logits /*Bx32*/,
                          const float* __restrict__ ob, float* __restrict__ out /*Bx26*/,
                          int Brows) {
    int r = blockIdx.x * blockDim.x + threadIdx.x;
    if (r >= Brows) return;
    const float* lr = logits + (long)r * 32;
    float z[26], mx = -3.4e38f;
#pragma unroll
    for (int j = 0; j < 26; ++j) { z[j] = lr[j] + ob[j]; mx = fmaxf(mx, z[j]); }
    float s = 0.0f;
#pragma unroll
    for (int j = 0; j < 26; ++j) { z[j] = __expf(z[j] - mx); s += z[j]; }
    float inv = __fdividef(1.0f, s);
#pragma unroll
    for (int j = 0; j < 26; ++j) out[(long)r * 26 + j] = z[j] * inv;
}

// ---------------------------------------------------------------------------
// Host driver
// ---------------------------------------------------------------------------
static inline long cdiv(long a, long b) { return (a + b - 1) / b; }

extern "C" void kernel_launch(void* const* d_in, const int* in_sizes, int n_in,
                              void* d_out, int out_size, void* d_ws, size_t ws_size,
                              hipStream_t stream) {
    (void)in_sizes; (void)n_in; (void)out_size; (void)ws_size;
    const float* x    = (const float*)d_in[0];
    const float* w1f  = (const float*)d_in[1];
    const float* u1f  = (const float*)d_in[2];
    const float* b1f  = (const float*)d_in[3];
    const float* w1b  = (const float*)d_in[4];
    const float* u1b  = (const float*)d_in[5];
    const float* b1b  = (const float*)d_in[6];
    const float* w2f  = (const float*)d_in[7];
    const float* u2f  = (const float*)d_in[8];
    const float* b2f  = (const float*)d_in[9];
    const float* w2b  = (const float*)d_in[10];
    const float* u2b  = (const float*)d_in[11];
    const float* b2b  = (const float*)d_in[12];
    const float* w3f  = (const float*)d_in[13];
    const float* u3f  = (const float*)d_in[14];
    const float* b3f  = (const float*)d_in[15];
    const float* w3b  = (const float*)d_in[16];
    const float* u3b  = (const float*)d_in[17];
    const float* b3b  = (const float*)d_in[18];
    const float* d1w  = (const float*)d_in[19];
    const float* d1b  = (const float*)d_in[20];
    const float* d2w  = (const float*)d_in[21];
    const float* d2b  = (const float*)d_in[22];
    const float* d3w  = (const float*)d_in[23];
    const float* d3b  = (const float*)d_in[24];
    const float* bng  = (const float*)d_in[25];
    const float* bnb  = (const float*)d_in[26];
    const float* bnm  = (const float*)d_in[27];
    const float* bnv  = (const float*)d_in[28];
    const float* outw = (const float*)d_in[29];
    const float* outb = (const float*)d_in[30];

    char* ws = (char*)d_ws;
    size_t off = 0;
    auto take = [&](size_t bytes) -> char* {
        char* p = ws + off;
        off = (off + bytes + 255) & ~(size_t)255;
        return p;
    };
    f16* xpad   = (f16*)take((size_t)NBT * 128 * 2);          // x padded 126->128
    f16* wc1fT  = (f16*)take((size_t)256 * 192 * 2);          // (N4 x KTOT) transposed
    f16* wc1bT  = (f16*)take((size_t)256 * 192 * 2);
    f16* wc2fT  = (f16*)take((size_t)512 * 256 * 2);
    f16* wc2bT  = (f16*)take((size_t)512 * 256 * 2);
    f16* wc3fT  = (f16*)take((size_t)1024 * 512 * 2);
    f16* wc3bT  = (f16*)take((size_t)1024 * 512 * 2);
    f16* l1out  = (f16*)take((size_t)NBT * 128 * 2);
    f16* l2out  = (f16*)take((size_t)NBT * 256 * 2);
    f16* l3out  = (f16*)take((size_t)NBT * 512 * 2);
    f16* d1wT   = (f16*)take((size_t)256 * 15360 * 2);
    f16* d2wT   = (f16*)take((size_t)128 * 256 * 2);
    f16* d3wT   = (f16*)take((size_t)64 * 128 * 2);
    f16* owT    = (f16*)take((size_t)32 * 64 * 2);            // out_w^T, N-pad 26->32
    f16* d1o    = (f16*)take((size_t)BB * 256 * 2);
    f16* d2o    = (f16*)take((size_t)BB * 128 * 2);
    f16* d3o    = (f16*)take((size_t)BB * 64 * 2);
    f16* bno    = (f16*)take((size_t)BB * 64 * 2);
    float* lgt  = (float*)take((size_t)BB * 32 * 4);

    const int TPB = 256;
    auto zconv = [&](f16* d, long n) {
        zero_f16<<<dim3((unsigned)cdiv(n, TPB)), dim3(TPB), 0, stream>>>(d, n);
    };
    auto tconv = [&](const float* s, f16* d, int K, int N, int ld, int colOff) {
        long n = (long)K * N;
        convT_f16<<<dim3((unsigned)cdiv(n, TPB)), dim3(TPB), 0, stream>>>(s, d, K, N, ld, colOff);
    };

    // --- staging: x (row-padded), all weights transposed f16 ---
    {
        long n = (long)NBT * 128;
        pad_convert<<<dim3((unsigned)cdiv(n, TPB)), dim3(TPB), 0, stream>>>(x, xpad, NBT, 126, 128);
    }
    zconv(wc1fT, 256L * 192); zconv(wc1bT, 256L * 192);
    tconv(w1f, wc1fT, 126, 256, 192, 0);   tconv(u1f, wc1fT, 64, 256, 192, 128);
    tconv(w1b, wc1bT, 126, 256, 192, 0);   tconv(u1b, wc1bT, 64, 256, 192, 128);
    tconv(w2f, wc2fT, 128, 512, 256, 0);   tconv(u2f, wc2fT, 128, 512, 256, 128);
    tconv(w2b, wc2bT, 128, 512, 256, 0);   tconv(u2b, wc2bT, 128, 512, 256, 128);
    tconv(w3f, wc3fT, 256, 1024, 512, 0);  tconv(u3f, wc3fT, 256, 1024, 512, 256);
    tconv(w3b, wc3bT, 256, 1024, 512, 0);  tconv(u3b, wc3bT, 256, 1024, 512, 256);
    tconv(d1w, d1wT, 15360, 256, 15360, 0);
    tconv(d2w, d2wT, 256, 128, 256, 0);
    tconv(d3w, d3wT, 128, 64, 128, 0);
    zconv(owT, 32L * 64);
    tconv(outw, owT, 64, 26, 64, 0);

    // --- BiLSTM stack: 512 blocks x 8 waves ---
    const dim3 sgrid(BB / 16), sblk(256);
    constexpr size_t s1 = (size_t)256 * 192 * 2 + 352 * 64;    // 120832: weights in LDS
    constexpr size_t s2 = (size_t)512 * 256 * 2 + 352 * 128;   // 307200: weights in LDS
    constexpr size_t s3 = (size_t)352 * 256;                   //  90112: weights via L2
    lstm_scan<128, 64, true ><<<sgrid, sblk, s1, stream>>>(xpad,  wc1fT, b1f, l1out, 0, 0);
    lstm_scan<128, 64, true ><<<sgrid, sblk, s1, stream>>>(xpad,  wc1bT, b1b, l1out, 1, 1);
    lstm_scan<128, 128, true><<<sgrid, sblk, s2, stream>>>(l1out, wc2fT, b2f, l2out, 0, 0);
    lstm_scan<128, 128, true><<<sgrid, sblk, s2, stream>>>(l1out, wc2bT, b2b, l2out, 1, 1);
    lstm_scan<256, 256, false><<<sgrid, sblk, s3, stream>>>(l2out, wc3fT, b3f, l3out, 0, 0);
    lstm_scan<256, 256, false><<<sgrid, sblk, s3, stream>>>(l2out, wc3bT, b3b, l3out, 1, 1);

    // --- dense head (l3out viewed as (B x 15360) f16, matches reshape order) ---
    wmma_gemm_bias<4, true, true><<<dim3(BB / 16, 4), dim3(32), 0, stream>>>(
        l3out, d1wT, d1b, d1o, nullptr, BB, 256, 15360);
    wmma_gemm_bias<4, true, true><<<dim3(BB / 16, 2), dim3(32), 0, stream>>>(
        d1o, d2wT, d2b, d2o, nullptr, BB, 128, 256);
    wmma_gemm_bias<4, true, true><<<dim3(BB / 16, 1), dim3(32), 0, stream>>>(
        d2o, d3wT, d3b, d3o, nullptr, BB, 64, 128);
    {
        long n = (long)BB * 64;
        bn_kernel<<<dim3((unsigned)cdiv(n, TPB)), dim3(TPB), 0, stream>>>(
            d3o, bng, bnb, bnm, bnv, bno, n);
    }
    wmma_gemm_bias<2, false, false><<<dim3(BB / 16, 1), dim3(32), 0, stream>>>(
        bno, owT, nullptr, nullptr, lgt, BB, 32, 64);
    softmax26<<<dim3((unsigned)cdiv(BB, TPB)), dim3(TPB), 0, stream>>>(
        lgt, outb, (float*)d_out, BB);
}
